// MLPNetworkWithAttention1_80169859547540
// MI455X (gfx1250) — matrix operand order repaired and checked by reading, not compiled, not run on hardware
//
#include <hip/hip_runtime.h>
#include <hip/hip_bf16.h>

typedef __attribute__((ext_vector_type(16))) _Float16 v16h;
typedef __attribute__((ext_vector_type(8)))  _Float16 v8h;
typedef __attribute__((ext_vector_type(8)))  float    v8f;

#define BATCH   262144
#define NTILES  (BATCH / 16)
#define LRELU   0.01f

static __device__ __forceinline__ v16h cat8(v8h lo, v8h hi) {
  return __builtin_shufflevector(lo, hi, 0,1,2,3,4,5,6,7,8,9,10,11,12,13,14,15);
}

// ---------------------------------------------------------------------------
// k_prep: convert weights to f16 (embed weights padded 13 -> 32 so every lane
// of the WMMA B-fragment load is a valid, uniform 32B load), zero stats.
// Runs every call so the stats accumulator is deterministic.
// ---------------------------------------------------------------------------
__global__ __launch_bounds__(256) void k_prep(
    const float* __restrict__ We, const float* __restrict__ Wk1,
    const float* __restrict__ Wk2, const float* __restrict__ Wq,
    const float* __restrict__ Wk, const float* __restrict__ Wv,
    const float* __restrict__ W2,
    _Float16* __restrict__ WeP, _Float16* __restrict__ WqH,
    _Float16* __restrict__ W2h, float* __restrict__ stats)
{
  int tid = blockIdx.x * blockDim.x + threadIdx.x;
  int np  = gridDim.x * blockDim.x;
  for (int i = tid; i < 3 * 256 * 32; i += np) {        // [3][256][32], K padded
    int j = i >> 13, r = (i >> 5) & 255, k = i & 31;
    const float* s = (j == 0) ? We : (j == 1) ? Wk1 : Wk2;
    WeP[i] = (_Float16)((k < 13) ? s[r * 13 + k] : 0.f);
  }
  for (int i = tid; i < 3 * 65536; i += np) {
    const float* s = (i < 65536) ? Wq : (i < 131072) ? Wk : Wv;
    WqH[i] = (_Float16)s[i & 65535];
  }
  for (int i = tid; i < 128 * 256; i += np) W2h[i] = (_Float16)W2[i];
  for (int i = tid; i < 1024; i += np) stats[i] = 0.f;
}

// ---------------------------------------------------------------------------
// k_attn: embeddings + attention per 16-row batch tile, writes X=[B,512] f16
// and accumulates per-feature sum / sumsq into stats[1024] via atomics.
// 256 threads = 8 waves. WMMA f32_16x16x32_f16 everywhere.
// ---------------------------------------------------------------------------
__global__ __launch_bounds__(256) void k_attn(
    const float* __restrict__ x,
    const float* __restrict__ be, const float* __restrict__ bk1,
    const float* __restrict__ bk2, const float* __restrict__ bv,
    const _Float16* __restrict__ WeP, const _Float16* __restrict__ WqH,
    _Float16* __restrict__ X, float* __restrict__ stats)
{
  __shared__ __align__(16) _Float16 sFeat[3][16][16];   // padded 13->16
  __shared__ __align__(16) _Float16 sEmb[3][16][256];   // e_q, e_k1, e_k2
  __shared__ __align__(16) float    sQKV[5][16][256];   // Q,K1,K2,V1,V2
  __shared__ float sSc[2][16][16];
  __shared__ float sW[2][16];

  const int t = threadIdx.x;
  const int lane = t & 31, wave = t >> 5;
  const int M = lane & 15, hiL = lane >> 4, ks = hiL * 8;

  float sumA = 0.f, sqA = 0.f, sumB = 0.f, sqB = 0.f;

  for (int tile = blockIdx.x; tile < NTILES; tile += gridDim.x) {
    const int row0 = tile * 16;

    { // stage per-agent 13-dim features (zero padded)
      int m = t >> 4, k = t & 15;
      const float* xr = x + (size_t)(row0 + m) * 39;
      #pragma unroll
      for (int j = 0; j < 3; ++j) {
        float v = (k < 12) ? xr[12 * j + k] : ((k == 12) ? xr[36 + j] : 0.f);
        sFeat[j][m][k] = (_Float16)v;
      }
    }
    __syncthreads();

    // embeddings: 3 GEMMs [16x256], K=16 (zero-padded inside 16x16x32)
    for (int task = wave; task < 48; task += 8) {
      int e = task >> 4, nb = (task & 15) << 4;
      v8h zz = {};
      v8h lo = *(const v8h*)&sFeat[e][M][ks];
      v16h a = cat8(lo, zz);                                      // K>=16 zero
      // uniform load: WeP rows are 32 halves, K=13..31 are zeros
      v16h b = cat8(*(const v8h*)(WeP + e * 8192 + (size_t)(nb + M) * 32 + ks), *(const v8h*)(WeP + e * 8192 + (size_t)(nb + M) * 32 + 16 + ks));
      const float* bias = (e == 0) ? be : (e == 1) ? bk1 : bk2;
      float bc = bias[nb + M];
      v8f c;
      #pragma unroll
      for (int r = 0; r < 8; ++r) c[r] = bc;
      c = __builtin_amdgcn_wmma_f32_16x16x32_f16(false, a, false, b, (short)0, c, false, false);
      #pragma unroll
      for (int r = 0; r < 8; ++r)
        sEmb[e][r + hiL * 8][nb + M] = (_Float16)c[r];
    }
    __syncthreads();

    // projections: Q, K1, K2, V1, V2  ([16x256], K=256 -> 8 WMMA each tile)
    for (int task = wave; task < 80; task += 8) {
      int g = task >> 4, nb = (task & 15) << 4;
      int src = (g == 0) ? 0 : ((g == 1 || g == 3) ? 1 : 2);
      const _Float16* W = WqH + ((g == 0) ? 0 : (g < 3) ? 65536 : 131072);
      float bc = (g >= 3) ? bv[nb + M] : 0.f;
      v8f c;
      #pragma unroll
      for (int r = 0; r < 8; ++r) c[r] = bc;
      for (int kb = 0; kb < 256; kb += 32) {
        v8h lo = *(const v8h*)&sEmb[src][M][kb + ks];
        v8h hb = *(const v8h*)&sEmb[src][M][kb + 16 + ks];
        v16h a = cat8(lo, hb);
        v16h b = cat8(*(const v8h*)(W + (size_t)(nb + M) * 256 + kb + ks), *(const v8h*)(W + (size_t)(nb + M) * 256 + kb + 16 + ks));
        c = __builtin_amdgcn_wmma_f32_16x16x32_f16(false, a, false, b, (short)0, c, false, false);
      }
      #pragma unroll
      for (int r = 0; r < 8; ++r) {
        float v = c[r];
        if (g >= 3) v = (v > 0.f) ? v : LRELU * v;   // leaky_relu on V
        sQKV[g][r + hiL * 8][nb + M] = v;
      }
    }
    __syncthreads();

    { // scores: s_n = (Q . K_n) / 16
      int m = t >> 4, seg = t & 15;
      float p1 = 0.f, p2 = 0.f;
      #pragma unroll
      for (int j = 0; j < 16; ++j) {
        int c2 = seg * 16 + j;
        float q = sQKV[0][m][c2];
        p1 += q * sQKV[1][m][c2];
        p2 += q * sQKV[2][m][c2];
      }
      sSc[0][m][seg] = p1; sSc[1][m][seg] = p2;
    }
    __syncthreads();
    if (t < 16) { // softmax over n=2
      float s1 = 0.f, s2 = 0.f;
      #pragma unroll
      for (int j = 0; j < 16; ++j) { s1 += sSc[0][t][j]; s2 += sSc[1][t][j]; }
      s1 *= 0.0625f; s2 *= 0.0625f;
      float mx = fmaxf(s1, s2);
      float e1 = __expf(s1 - mx), e2 = __expf(s2 - mx);
      float inv = 1.f / (e1 + e2);
      sW[0][t] = e1 * inv; sW[1][t] = e2 * inv;
    }
    __syncthreads();

    { // X writes (both halves, coalesced: thread t owns feature column t)
      // + running batch stats
      #pragma unroll 4
      for (int m = 0; m < 16; ++m) {
        _Float16* xrow = X + (size_t)(row0 + m) * 512;
        _Float16 eqh = sEmb[0][m][t];
        xrow[t] = eqh;                                   // X[:,0:256] = e_q
        float eq = (float)eqh;
        sumA += eq; sqA += eq * eq;
        float a = sW[0][m] * sQKV[3][m][t] + sW[1][m] * sQKV[4][m][t];
        xrow[256 + t] = (_Float16)a;                     // X[:,256:512] = atten
        sumB += a; sqB += a * a;
      }
    }
    __syncthreads();
  }
  atomicAdd(stats + t,       sumA);   // sum   of feature t
  atomicAdd(stats + 256 + t, sumB);   // sum   of feature 256+t
  atomicAdd(stats + 512 + t, sqA);    // sumsq of feature t
  atomicAdd(stats + 768 + t, sqB);    // sumsq of feature 256+t
}

// ---------------------------------------------------------------------------
// k_bnfold: finalize mean / rsqrt(var+eps) and fold BN into W1,b1.
// ---------------------------------------------------------------------------
__global__ __launch_bounds__(512) void k_bnfold(
    const float* __restrict__ stats, const float* __restrict__ W1,
    const float* __restrict__ b1,
    _Float16* __restrict__ W1p, float* __restrict__ b1p)
{
  __shared__ float sMean[512], sInv[512];
  int t = threadIdx.x;
  {
    float mean = stats[t] * (1.f / (float)BATCH);
    float var  = stats[512 + t] * (1.f / (float)BATCH) - mean * mean;
    sMean[t] = mean;
    sInv[t]  = rsqrtf(var + 1e-5f);
  }
  __syncthreads();
  for (int i = t; i < 256 * 512; i += 512) {
    int ii = i & 511;
    W1p[i] = (_Float16)(W1[i] * sInv[ii]);
  }
  if (t < 256) {
    float acc = b1[t];
    const float* wr = W1 + (size_t)t * 512;
    for (int i = 0; i < 512; ++i) acc -= wr[i] * sInv[i] * sMean[i];
    b1p[t] = acc;
  }
}

// ---------------------------------------------------------------------------
// k_mlp: per 16-row tile: relu(X @ W1p^T + b1p) -> relu(. @ W2^T + b2) -> W3.
// ---------------------------------------------------------------------------
__global__ __launch_bounds__(256) void k_mlp(
    const _Float16* __restrict__ X,
    const _Float16* __restrict__ W1p, const float* __restrict__ b1p,
    const _Float16* __restrict__ W2h, const float* __restrict__ b2,
    const float* __restrict__ W3, const float* __restrict__ b3,
    float* __restrict__ out)
{
  __shared__ __align__(16) _Float16 sH1[16][256];
  __shared__ __align__(16) float    sH2[16][128];
  __shared__ float sRed[16][16];

  const int t = threadIdx.x;
  const int lane = t & 31, wave = t >> 5;
  const int M = lane & 15, hiL = lane >> 4, ks = hiL * 8;
  const int row0 = blockIdx.x * 16;

  // layer 1: [16 x 256], K = 512 (BN folded into W1p/b1p)
  for (int task = wave; task < 16; task += 8) {
    int nb = task << 4;
    float bc = b1p[nb + M];
    v8f c;
    #pragma unroll
    for (int r = 0; r < 8; ++r) c[r] = bc;
    const _Float16* xr = X + (size_t)(row0 + M) * 512;
    for (int kb = 0; kb < 512; kb += 32) {
      v8h lo = *(const v8h*)(xr + kb + ks);
      v8h hb = *(const v8h*)(xr + kb + 16 + ks);
      v16h a = cat8(lo, hb);
      v16h b = cat8(*(const v8h*)(W1p + (size_t)(nb + M) * 512 + kb + ks), *(const v8h*)(W1p + (size_t)(nb + M) * 512 + kb + 16 + ks));
      c = __builtin_amdgcn_wmma_f32_16x16x32_f16(false, a, false, b, (short)0, c, false, false);
    }
    #pragma unroll
    for (int r = 0; r < 8; ++r)
      sH1[r + hiL * 8][nb + M] = (_Float16)fmaxf(c[r], 0.f);
  }
  __syncthreads();

  // layer 2: [16 x 128], K = 256 (one 16x16 tile per wave)
  {
    int nb = wave << 4;
    float bc = b2[nb + M];
    v8f c;
    #pragma unroll
    for (int r = 0; r < 8; ++r) c[r] = bc;
    for (int kb = 0; kb < 256; kb += 32) {
      v8h lo = *(const v8h*)&sH1[M][kb + ks];
      v8h hb = *(const v8h*)&sH1[M][kb + 16 + ks];
      v16h a = cat8(lo, hb);
      v16h b = cat8(*(const v8h*)(W2h + (size_t)(nb + M) * 256 + kb + ks), *(const v8h*)(W2h + (size_t)(nb + M) * 256 + kb + 16 + ks));
      c = __builtin_amdgcn_wmma_f32_16x16x32_f16(false, a, false, b, (short)0, c, false, false);
    }
    #pragma unroll
    for (int r = 0; r < 8; ++r)
      sH2[r + hiL * 8][nb + M] = fmaxf(c[r], 0.f);
  }
  __syncthreads();

  // layer 3: [16 x 1] dot with W3 (VALU reduce)
  {
    int m = t >> 4, seg = t & 15;
    float p = 0.f;
    #pragma unroll
    for (int j = 0; j < 8; ++j) p += sH2[m][seg * 8 + j] * W3[seg * 8 + j];
    sRed[m][seg] = p;
  }
  __syncthreads();
  if (t < 16) {
    float s = 0.f;
    #pragma unroll
    for (int j = 0; j < 16; ++j) s += sRed[t][j];
    out[row0 + t] = s + b3[0];
  }
}

// ---------------------------------------------------------------------------
extern "C" void kernel_launch(void* const* d_in, const int* in_sizes, int n_in,
                              void* d_out, int out_size, void* d_ws, size_t ws_size,
                              hipStream_t stream) {
  (void)in_sizes; (void)n_in; (void)out_size; (void)ws_size;
  const float* x   = (const float*)d_in[0];
  const float* We  = (const float*)d_in[1];
  const float* be  = (const float*)d_in[2];
  const float* Wk1 = (const float*)d_in[3];
  const float* bk1 = (const float*)d_in[4];
  const float* Wk2 = (const float*)d_in[5];
  const float* bk2 = (const float*)d_in[6];
  const float* Wq  = (const float*)d_in[7];
  const float* Wk  = (const float*)d_in[8];
  const float* Wv  = (const float*)d_in[9];
  const float* bv  = (const float*)d_in[10];
  const float* W1  = (const float*)d_in[11];
  const float* b1  = (const float*)d_in[12];
  const float* W2  = (const float*)d_in[13];
  const float* b2  = (const float*)d_in[14];
  const float* W3  = (const float*)d_in[15];
  const float* b3  = (const float*)d_in[16];
  float* out = (float*)d_out;

  char* ws = (char*)d_ws;
  size_t o = 0;
  _Float16* X    = (_Float16*)(ws + o); o += (size_t)BATCH * 512 * 2;   // 256 MB
  _Float16* WeP  = (_Float16*)(ws + o); o += 3 * 256 * 32 * 2;          // K padded to 32
  _Float16* WqH  = (_Float16*)(ws + o); o += 3 * 256 * 256 * 2;
  _Float16* W1p  = (_Float16*)(ws + o); o += 256 * 512 * 2;
  _Float16* W2h  = (_Float16*)(ws + o); o += 128 * 256 * 2;
  float*    b1p  = (float*)(ws + o);    o += 256 * 4;
  float*    stats= (float*)(ws + o);    o += 1024 * 4;

  k_prep  <<<256,    256, 0, stream>>>(We, Wk1, Wk2, Wq, Wk, Wv, W2, WeP, WqH, W2h, stats);
  k_attn  <<<2048,   256, 0, stream>>>(x, be, bk1, bk2, bv, WeP, WqH, X, stats);
  k_bnfold<<<1,      512, 0, stream>>>(stats, W1, b1, W1p, b1p);
  k_mlp   <<<NTILES, 256, 0, stream>>>(X, W1p, b1p, W2h, b2, W3, b3, out);
}
